// GNN_2061584302171
// MI455X (gfx1250) — matrix operand order repaired and checked by reading, not compiled, not run on hardware
//
#include <hip/hip_runtime.h>

#define NN      50000          // nodes
#define DF      256            // feature dim (in = hid)
#define NCLS    16             // classes
#define NT      (NN / 16)      // 3125 row tiles (exact)

typedef __attribute__((ext_vector_type(16))) __bf16 v16bf;
typedef __attribute__((ext_vector_type(8)))  float  v8f;
typedef __attribute__((ext_vector_type(4)))  unsigned int v4u;
typedef __attribute__((ext_vector_type(8)))  int    v8i;
typedef __attribute__((ext_vector_type(4)))  int    v4i;

union BFrag {                  // 16 bf16 = 32 bytes
    unsigned short s[16];
    uint4          u4[2];
    v16bf          v;
};
union AccFrag {
    v8f   v;
    float f[8];
};

__device__ __forceinline__ unsigned short f2bf(float f) {
    unsigned u = __float_as_uint(f);
    return (unsigned short)((u + 0x7FFFu + ((u >> 16) & 1u)) >> 16); // RNE
}

// ---------------- degree / norm ----------------
__global__ void deg_init_kernel(float* deg) {
    int i = blockIdx.x * blockDim.x + threadIdx.x;
    if (i < NN) deg[i] = 1.0f;                       // self-loop
}
__global__ void deg_acc_kernel(const long long* __restrict__ dst, float* deg, int E) {
    int e = blockIdx.x * blockDim.x + threadIdx.x;
    if (e < E) atomicAdd(&deg[(int)dst[e]], 1.0f);
}
__global__ void dinv_kernel(float* deg) {
    int i = blockIdx.x * blockDim.x + threadIdx.x;
    if (i < NN) deg[i] = rsqrtf(deg[i]);             // deg>0 always (self-loop)
}
__global__ void norm_kernel(const long long* __restrict__ src,
                            const long long* __restrict__ dst,
                            const float* __restrict__ dinv,
                            float* __restrict__ norm, int E) {
    int e = blockIdx.x * blockDim.x + threadIdx.x;
    if (e < E) norm[e] = dinv[(int)src[e]] * dinv[(int)dst[e]];
}

// ---------------- weight repack: fp32 -> bf16 B-fragment layout ----------------
// out[((ct*16 + kblk)*16 + col)*16 + kk] = bf16(W[(kblk*16+kk)*ldw + ct*16+col])
__global__ void packW1_kernel(const float* __restrict__ W, unsigned short* __restrict__ o) {
    int idx = blockIdx.x * blockDim.x + threadIdx.x;      // 65536
    int kk = idx & 15, col = (idx >> 4) & 15, kblk = (idx >> 8) & 15, ct = idx >> 12;
    o[idx] = f2bf(W[(kblk * 16 + kk) * DF + ct * 16 + col]);
}
__global__ void packW2_kernel(const float* __restrict__ W, unsigned short* __restrict__ o) {
    int idx = blockIdx.x * blockDim.x + threadIdx.x;      // 4096
    int kk = idx & 15, col = (idx >> 4) & 15, kblk = idx >> 8;
    o[idx] = f2bf(W[(kblk * 16 + kk) * NCLS + col]);
}

// ---------------- GEMM1: H[50000,256] = X @ W1, TDM-staged A + bf16 WMMA ----------
__global__ void gemm1_kernel(const float* __restrict__ X,
                             const unsigned short* __restrict__ Wb,
                             float* __restrict__ H) {
    __shared__ __align__(16) float          sXf[16 * DF]; // 16 KB fp32 tile via TDM
    __shared__ __align__(16) unsigned short sA[16 * DF];  // bf16 A-fragment staging
    const int tile = blockIdx.x;                          // 0..3124

    // --- Tensor Data Mover: DMA the 16x256 fp32 tile into LDS (wave 0 only; TDM
    // ignores EXEC and issues per wave, tracked on TENSORcnt). Descriptors per
    // ISA 8.3/8.4: 2-D tensor, data_size=4B, tile 256x16, stride 256 elements.
    if (threadIdx.x < 32) {
        unsigned long long ga = (unsigned long long)(X + (size_t)tile * 16 * DF);
        unsigned ldsOff = (unsigned)(size_t)(void*)sXf;   // low 32 bits = LDS byte offset
        v4u g0;
        g0[0] = 1u;                                       // count=1, user mode
        g0[1] = ldsOff;                                   // lds_addr
        g0[2] = (unsigned)(ga & 0xFFFFFFFFull);           // global_addr[31:0]
        g0[3] = (unsigned)((ga >> 32) & 0x01FFFFFFull)    // global_addr[56:32]
              | 0x80000000u;                              // type=2 ("image")
        v8i g1;
        g1[0] = (int)(2u << 16);                          // data_size=4B, no mask/flags
        g1[1] = (int)((DF & 0xFFFFu) << 16);              // tensor_dim0 = 256 (lo16)
        g1[2] = (int)(16u << 16);                         // tensor_dim0 hi=0 | tensor_dim1=16 lo
        g1[3] = (int)((unsigned)DF << 16);                // tensor_dim1 hi=0 | tile_dim0=256
        g1[4] = 16;                                       // tile_dim1=16, tile_dim2=0
        g1[5] = DF;                                       // tensor_dim0_stride = 256 (lo32)
        g1[6] = 0;                                        // stride hi | tensor_dim1_stride lo
        g1[7] = 0;
        v4i gz4 = {0, 0, 0, 0};                           // 2-D: groups 2/3 unused
        v8i gz8 = {0, 0, 0, 0, 0, 0, 0, 0};
        __builtin_amdgcn_tensor_load_to_lds(g0, g1, gz4, gz4, gz8, 0);
        __builtin_amdgcn_s_wait_tensorcnt(0);
    }
    __syncthreads();

    // fp32 -> bf16 conversion pass in LDS (vector mix, one shot)
    for (int i = threadIdx.x; i < 16 * DF; i += 256) sA[i] = f2bf(sXf[i]);
    __syncthreads();

    const int wave  = threadIdx.x >> 5;
    const int lane  = threadIdx.x & 31;
    const int m     = lane & 15;
    const int khalf = lane >> 4;        // 0: K {0..15}  1: K {16..31}
    const int col   = lane & 15;
    const int ct0   = wave * 2, ct1 = ct0 + 1;

    AccFrag acc0, acc1;
    acc0.v = (v8f){}; acc1.v = (v8f){};
    for (int k0 = 0; k0 < DF; k0 += 32) {
        BFrag a, b0, b1;
        const int kb = k0 + khalf * 16;
        a.u4[0] = *(const uint4*)&sA[m * DF + kb];
        a.u4[1] = *(const uint4*)&sA[m * DF + kb + 8];
        const int kblk = (k0 >> 4) + khalf;
        const unsigned short* p0 = Wb + ((((ct0 << 4) + kblk) << 4) + col) * 16;
        const unsigned short* p1 = Wb + ((((ct1 << 4) + kblk) << 4) + col) * 16;
        b0.u4[0] = *(const uint4*)p0; b0.u4[1] = *(const uint4*)(p0 + 8);
        b1.u4[0] = *(const uint4*)p1; b1.u4[1] = *(const uint4*)(p1 + 8);
        acc0.v = __builtin_amdgcn_wmma_f32_16x16x32_bf16(false, a.v, false, b0.v,
                                                         (short)0, acc0.v, false, false);
        acc1.v = __builtin_amdgcn_wmma_f32_16x16x32_bf16(false, a.v, false, b1.v,
                                                         (short)0, acc1.v, false, false);
    }
    const int rowBase = tile * 16 + khalf * 8;            // D layout: VGPR r -> M=r / M=8+r
#pragma unroll
    for (int r = 0; r < 8; ++r) {
        H[(size_t)(rowBase + r) * DF + ct0 * 16 + col] = acc0.f[r];
        H[(size_t)(rowBase + r) * DF + ct1 * 16 + col] = acc1.f[r];
    }
}

// ---------------- GEMM2: H2[50000,16] = A @ W2, bf16 WMMA ----------------
__global__ void gemm2_kernel(const float* __restrict__ A,
                             const unsigned short* __restrict__ Wb2,
                             float* __restrict__ H2) {
    const int rowTile = blockIdx.x * 8 + (threadIdx.x >> 5);
    if (rowTile >= NT) return;                            // wave-uniform guard
    const int lane  = threadIdx.x & 31;
    const int m     = lane & 15;
    const int khalf = lane >> 4;
    const int col   = lane & 15;
    const float* arow = A + ((size_t)rowTile * 16 + m) * DF;

    AccFrag acc; acc.v = (v8f){};
    for (int k0 = 0; k0 < DF; k0 += 32) {
        const int kb = k0 + khalf * 16;
        __builtin_prefetch(arow + kb + 32, 0, 0);
        float4 f0 = *(const float4*)(arow + kb);
        float4 f1 = *(const float4*)(arow + kb + 4);
        float4 f2 = *(const float4*)(arow + kb + 8);
        float4 f3 = *(const float4*)(arow + kb + 12);
        BFrag a, b;
        a.s[0]  = f2bf(f0.x); a.s[1]  = f2bf(f0.y); a.s[2]  = f2bf(f0.z); a.s[3]  = f2bf(f0.w);
        a.s[4]  = f2bf(f1.x); a.s[5]  = f2bf(f1.y); a.s[6]  = f2bf(f1.z); a.s[7]  = f2bf(f1.w);
        a.s[8]  = f2bf(f2.x); a.s[9]  = f2bf(f2.y); a.s[10] = f2bf(f2.z); a.s[11] = f2bf(f2.w);
        a.s[12] = f2bf(f3.x); a.s[13] = f2bf(f3.y); a.s[14] = f2bf(f3.z); a.s[15] = f2bf(f3.w);
        const int kblk = (k0 >> 4) + khalf;
        const unsigned short* p = Wb2 + ((kblk << 4) + col) * 16;
        b.u4[0] = *(const uint4*)p; b.u4[1] = *(const uint4*)(p + 8);
        acc.v = __builtin_amdgcn_wmma_f32_16x16x32_bf16(false, a.v, false, b.v,
                                                        (short)0, acc.v, false, false);
    }
    const int rowBase = rowTile * 16 + khalf * 8;
#pragma unroll
    for (int r = 0; r < 8; ++r)
        H2[(size_t)(rowBase + r) * NCLS + col] = acc.f[r];
}

// ---------------- aggregation ----------------
// init AGG with self-loop contribution: AGG[i] = H[i] * dinv[i]^2  (also zero-fills)
__global__ void selfloop_init_kernel(const float* __restrict__ H,
                                     const float* __restrict__ dinv,
                                     float* __restrict__ AGG, int quads) {
    long long t = (long long)blockIdx.x * blockDim.x + threadIdx.x;
    if (t >= (long long)NN * quads) return;
    int node = (int)(t / quads), q = (int)(t % quads);
    float w = dinv[node]; w *= w;
    const int width = quads * 4;
    float4 v = *(const float4*)(H + (size_t)node * width + q * 4);
    v.x *= w; v.y *= w; v.z *= w; v.w *= w;
    *(float4*)(AGG + (size_t)node * width + q * 4) = v;
}

__global__ void scatter_kernel(const long long* __restrict__ src,
                               const long long* __restrict__ dst,
                               const float* __restrict__ norm,
                               const float* __restrict__ H,
                               float* __restrict__ AGG, int E, int quads) {
    long long t = (long long)blockIdx.x * blockDim.x + threadIdx.x;
    if (t >= (long long)E * quads) return;
    int e = (int)(t / quads), q = (int)(t % quads);
    int s = (int)src[e], d = (int)dst[e];
    float w = norm[e];
    const int width = quads * 4;
    float4 v = *(const float4*)(H + (size_t)s * width + q * 4);
    float* op = AGG + (size_t)d * width + q * 4;
    atomicAdd(op + 0, v.x * w);
    atomicAdd(op + 1, v.y * w);
    atomicAdd(op + 2, v.z * w);
    atomicAdd(op + 3, v.w * w);
}

__global__ void bias_relu_kernel(float* __restrict__ A, const float* __restrict__ b) {
    long long t = (long long)blockIdx.x * blockDim.x + threadIdx.x;
    if (t >= (long long)NN * (DF / 4)) return;
    int c4 = (int)(t % (DF / 4));
    float4 v = *(float4*)(A + t * 4);
    v.x = fmaxf(v.x + b[c4 * 4 + 0], 0.0f);
    v.y = fmaxf(v.y + b[c4 * 4 + 1], 0.0f);
    v.z = fmaxf(v.z + b[c4 * 4 + 2], 0.0f);
    v.w = fmaxf(v.w + b[c4 * 4 + 3], 0.0f);
    *(float4*)(A + t * 4) = v;
}

__global__ void softmax_kernel(const float* __restrict__ A, const float* __restrict__ b2,
                               float* __restrict__ out) {
    int r = blockIdx.x * blockDim.x + threadIdx.x;
    if (r >= NN) return;
    float v[NCLS];
    float mx = -3.402823466e38f;
#pragma unroll
    for (int i = 0; i < NCLS; ++i) { v[i] = A[(size_t)r * NCLS + i] + b2[i]; mx = fmaxf(mx, v[i]); }
    float sum = 0.0f;
#pragma unroll
    for (int i = 0; i < NCLS; ++i) { v[i] = __expf(v[i] - mx); sum += v[i]; }
    float inv = 1.0f / sum;
#pragma unroll
    for (int i = 0; i < NCLS; ++i) out[(size_t)r * NCLS + i] = v[i] * inv;
}

// ---------------- launch ----------------
extern "C" void kernel_launch(void* const* d_in, const int* in_sizes, int n_in,
                              void* d_out, int out_size, void* d_ws, size_t ws_size,
                              hipStream_t stream) {
    const float*     X    = (const float*)d_in[0];
    const long long* eidx = (const long long*)d_in[1];   // [2, E] int64
    const float*     W1   = (const float*)d_in[2];
    const float*     b1   = (const float*)d_in[3];
    const float*     W2   = (const float*)d_in[4];
    const float*     b2   = (const float*)d_in[5];
    float*           out  = (float*)d_out;

    const int E = in_sizes[1] / 2;
    const long long* src = eidx;
    const long long* dst = eidx + E;

    // workspace carve-up (all offsets 16B aligned)
    char* w = (char*)d_ws;
    size_t off = 0;
    float* dinv = (float*)(w + off);           off += (size_t)NN * 4;          // 200000
    unsigned short* w1b = (unsigned short*)(w + off); off += (size_t)DF * DF * 2;
    unsigned short* w2b = (unsigned short*)(w + off); off += (size_t)DF * NCLS * 2;
    float* norm = (float*)(w + off);           off += (size_t)E * 4;
    float* h1   = (float*)(w + off);           off += (size_t)NN * DF * 4;
    float* agg1 = (float*)(w + off);           off += (size_t)NN * DF * 4;
    float* h2   = (float*)(w + off);           off += (size_t)NN * NCLS * 4;
    float* agg2 = (float*)(w + off);

    const int B = 256;
    // degrees & norms
    deg_init_kernel<<<(NN + B - 1) / B, B, 0, stream>>>(dinv);
    deg_acc_kernel<<<(E + B - 1) / B, B, 0, stream>>>(dst, dinv, E);
    dinv_kernel<<<(NN + B - 1) / B, B, 0, stream>>>(dinv);
    norm_kernel<<<(E + B - 1) / B, B, 0, stream>>>(src, dst, dinv, norm, E);
    // weight repack to bf16 fragment layout
    packW1_kernel<<<(DF * DF) / B, B, 0, stream>>>(W1, w1b);
    packW2_kernel<<<(DF * NCLS) / B, B, 0, stream>>>(W2, w2b);
    // layer 1
    gemm1_kernel<<<NT, B, 0, stream>>>(X, w1b, h1);
    selfloop_init_kernel<<<(int)(((long long)NN * 64 + B - 1) / B), B, 0, stream>>>(h1, dinv, agg1, 64);
    scatter_kernel<<<(int)(((long long)E * 64 + B - 1) / B), B, 0, stream>>>(src, dst, norm, h1, agg1, E, 64);
    bias_relu_kernel<<<(int)(((long long)NN * 64 + B - 1) / B), B, 0, stream>>>(agg1, b1);
    // layer 2
    gemm2_kernel<<<(NT + 7) / 8, B, 0, stream>>>(agg1, w2b, h2);
    selfloop_init_kernel<<<(int)(((long long)NN * 4 + B - 1) / B), B, 0, stream>>>(h2, dinv, agg2, 4);
    scatter_kernel<<<(int)(((long long)E * 4 + B - 1) / B), B, 0, stream>>>(src, dst, norm, h2, agg2, E, 4);
    softmax_kernel<<<(NN + B - 1) / B, B, 0, stream>>>(agg2, b2, out);
}
